// MultiHeadAttention_77833397338481
// MI455X (gfx1250) — matrix operand order repaired and checked by reading, not compiled, not run on hardware
//
#include <hip/hip_runtime.h>
#include <hip/hip_bf16.h>
#include <math.h>

typedef __attribute__((ext_vector_type(16))) _Float16 v16h;
typedef __attribute__((ext_vector_type(8)))  float    v8f;
typedef __attribute__((ext_vector_type(4)))  unsigned v4u;
typedef __attribute__((ext_vector_type(8)))  int      v8i;
typedef __attribute__((ext_vector_type(4)))  int      v4i;

#define D_MODEL 1024
#define NHEADS  64     // reference "head" axis = d_head = 64
#define EHEAD   16     // per-head dim = num_heads = 16
#define BATCH   2
#define SEQ     1024
#define ROWS    (BATCH * SEQ)   // 2048

// low 32 bits of a flat shared-aperture pointer == LDS byte offset
static __device__ __forceinline__ unsigned lds_off(const void* p) {
  return (unsigned)(size_t)p;
}

// ---- async global->LDS 16B copy (ASYNCcnt path) ----
static __device__ __forceinline__ void async_copy16(unsigned ldsoff, const void* g) {
  asm volatile("global_load_async_to_lds_b128 %0, %1, off"
               :: "v"(ldsoff), "v"((unsigned long long)(size_t)g)
               : "memory");
}
static __device__ __forceinline__ void wait_async0() {
  asm volatile("s_wait_asynccnt 0" ::: "memory");
}

// ---- Tensor Data Mover: load 128x32 f16 tile (row stride tw elems) into LDS,
//      padding each 64B row with 8B so LDS row stride is 36 halves ----
#if __has_builtin(__builtin_amdgcn_tensor_load_to_lds)
#define HAVE_TDM 1
#else
#define HAVE_TDM 0
#endif

static __device__ __forceinline__ void tdm_load_tile_128x32(
    unsigned lds_byte, const _Float16* gsrc, unsigned tw, unsigned th) {
#if HAVE_TDM
  unsigned long long ga = (unsigned long long)(size_t)gsrc;
  v4u g0;
  g0[0] = 1u;                                            // count=1 (valid user D#)
  g0[1] = lds_byte;                                      // lds_addr
  g0[2] = (unsigned)(ga & 0xFFFFFFFFu);                  // global_addr[31:0]
  g0[3] = (unsigned)((ga >> 32) & 0x01FFFFFFu) | (2u << 30);  // addr[56:32] | type=2
  v8i g1;
  g1[0] = (int)((1u << 16)        // data_size = 2B
              | (1u << 20)        // pad_enable
              | (3u << 22)        // pad_interval: 16 DWORDs
              | (1u << 25));      // pad_amount: 2 DWORDs
  g1[1] = (int)((tw & 0xFFFFu) << 16);                               // tensor_dim0 lo
  g1[2] = (int)(((tw >> 16) & 0xFFFFu) | ((th & 0xFFFFu) << 16));    // dim0 hi | dim1 lo
  g1[3] = (int)(((th >> 16) & 0xFFFFu) | (32u << 16));               // dim1 hi | tile_dim0=32
  g1[4] = (int)128;                                                  // tile_dim1=128, tile_dim2=0
  g1[5] = (int)tw;                                                   // tensor_dim0_stride[31:0]
  g1[6] = 0;
  g1[7] = 0;
  v4i gz; gz[0] = gz[1] = gz[2] = gz[3] = 0;
#if defined(__clang_major__) && (__clang_major__ >= 23)
  v8i gz8; gz8[0]=gz8[1]=gz8[2]=gz8[3]=gz8[4]=gz8[5]=gz8[6]=gz8[7]=0;
  __builtin_amdgcn_tensor_load_to_lds(g0, g1, gz, gz, gz8, 0);
#else
  __builtin_amdgcn_tensor_load_to_lds(g0, g1, gz, gz, 0);
#endif
#else
  (void)lds_byte; (void)gsrc; (void)tw; (void)th;
#endif
}

// ---- row reductions within 16-lane groups of the WMMA C layout ----
// xor masks 1/2/4/8 never cross a 16-lane group, so a full-wave ds_swizzle
// with immediate pattern (and=0x1f, or=0, xor=m) == __shfl_xor(x, m, 16),
// with zero index-VALU and no address VGPRs.
#define SWZ_XOR(x, pat) \
  __int_as_float(__builtin_amdgcn_ds_swizzle(__float_as_int(x), (pat)))

static __device__ __forceinline__ float rowmax16(float x) {
  x = fmaxf(x, SWZ_XOR(x, 0x7C01));
  x = fmaxf(x, SWZ_XOR(x, 0x7C02));
  x = fmaxf(x, SWZ_XOR(x, 0x7C04));
  x = fmaxf(x, SWZ_XOR(x, 0x7C08));
  return x;
}
static __device__ __forceinline__ float rowsum16(float x) {
  x += SWZ_XOR(x, 0x7C01);
  x += SWZ_XOR(x, 0x7C02);
  x += SWZ_XOR(x, 0x7C04);
  x += SWZ_XOR(x, 0x7C08);
  return x;
}

// ---------------- f32 -> f16 conversion (streaming: NT loads) ----------------
__global__ void cvt_f32_f16(const float* __restrict__ src,
                            _Float16* __restrict__ dst, int n) {
  int i = blockIdx.x * blockDim.x + threadIdx.x;
  int stride = gridDim.x * blockDim.x;
  for (; i < n; i += stride)
    dst[i] = (_Float16)__builtin_nontemporal_load(&src[i]);
}

// ---------------- TDM double-buffered WMMA GEMM: out = A @ W^T + b ----------------
// A: f16 [2048,1024]; W: f16 [1024,1024]; HEAD_OUT: f16 [b,h,s,16] else f32 [2048,1024]
template <bool HEAD_OUT>
__global__ __launch_bounds__(256)
void gemm_wmma_tdm(const _Float16* __restrict__ A, const _Float16* __restrict__ W,
                   const float* __restrict__ bias, void* __restrict__ outp) {
  __shared__ _Float16 sA[2][128][36];
  __shared__ _Float16 sB[2][128][36];

  const int tid  = threadIdx.x;
  const int wave = tid >> 5, lane = tid & 31, l15 = lane & 15, half = lane >> 4;
  const int m0 = blockIdx.x * 128;
  const int n0 = blockIdx.y * 128;

  v8f acc[8];
#pragma unroll
  for (int ns = 0; ns < 8; ++ns)
#pragma unroll
    for (int v = 0; v < 8; ++v) acc[ns][v] = 0.0f;

#if HAVE_TDM
  if (wave == 0) {   // prologue DMA for buffer 0 (TDM ignores EXEC; 1 issue/wave)
    tdm_load_tile_128x32(lds_off(&sA[0][0][0]), A + (size_t)m0 * D_MODEL, D_MODEL, ROWS);
    tdm_load_tile_128x32(lds_off(&sB[0][0][0]), W + (size_t)n0 * D_MODEL, D_MODEL, D_MODEL);
  }
#endif

  for (int it = 0; it < D_MODEL / 32; ++it) {
    const int cur = it & 1;
#if HAVE_TDM
    if (wave == 0) {
      if (it + 1 < D_MODEL / 32) {   // lookahead DMA into the other buffer
        const int nxt = cur ^ 1;
        const int kt1 = (it + 1) * 32;
        tdm_load_tile_128x32(lds_off(&sA[nxt][0][0]),
                             A + (size_t)m0 * D_MODEL + kt1, D_MODEL, ROWS);
        tdm_load_tile_128x32(lds_off(&sB[nxt][0][0]),
                             W + (size_t)n0 * D_MODEL + kt1, D_MODEL, D_MODEL);
        __builtin_amdgcn_s_wait_tensorcnt((short)2);  // current tiles complete
      } else {
        __builtin_amdgcn_s_wait_tensorcnt((short)0);
      }
    }
#else
    {  // fallback: manual cooperative staging
      const int kt = it * 32;
      for (int i = tid; i < 128 * 32; i += 256) {
        int r = i >> 5, c = i & 31;
        sA[cur][r][c] = A[(size_t)(m0 + r) * D_MODEL + kt + c];
        sB[cur][r][c] = W[(size_t)(n0 + r) * D_MODEL + kt + c];
      }
    }
#endif
    __syncthreads();   // DMA'd tiles visible to all waves

    // A fragment: lanes 0-15 = M, half selects K group {0-7,16-23}/{8-15,24-31}
    v16h a;
    {
      const int rowA = wave * 16 + l15;
#pragma unroll
      for (int i = 0; i < 16; ++i) {
        int k = i + 8 * half + ((i >= 8) ? 8 : 0);
        a[i] = sA[cur][rowA][k];
      }
    }
    // Pre-load ALL B fragments, then issue the WMMA burst back-to-back:
    // one dscnt drain instead of a wait before every WMMA.
    v16h bfr[8];
#pragma unroll
    for (int ns = 0; ns < 8; ++ns) {
      const int rowB = ns * 16 + l15;
#pragma unroll
      for (int i = 0; i < 16; ++i) bfr[ns][i] = sB[cur][rowB][i + 8 * half + ((i >= 8) ? 8 : 0)];
    }
#pragma unroll
    for (int ns = 0; ns < 8; ++ns) {
      acc[ns] = __builtin_amdgcn_wmma_f32_16x16x32_f16(
          false, a, false, bfr[ns], (short)0, acc[ns], false, false);
    }
    __syncthreads();   // all waves done before buffer is overwritten
  }

#pragma unroll
  for (int ns = 0; ns < 8; ++ns) {
    int col = n0 + ns * 16 + l15;
    float bcol = bias[col];
    if (HEAD_OUT) {
      _Float16* outHead = (_Float16*)outp;
      int hh = col >> 4, e = col & 15;
#pragma unroll
      for (int v = 0; v < 8; ++v) {
        int row = m0 + wave * 16 + v + 8 * half;
        int bb = row >> 10, s = row & 1023;
        outHead[(((size_t)bb * NHEADS + hh) * SEQ + s) * EHEAD + e] =
            (_Float16)(acc[ns][v] + bcol);
      }
    } else {
      float* out = (float*)outp;
#pragma unroll
      for (int v = 0; v < 8; ++v) {
        int row = m0 + wave * 16 + v + 8 * half;
        // final output is never re-read: stream it past the near caches
        __builtin_nontemporal_store(acc[ns][v] + bcol, &out[(size_t)row * D_MODEL + col]);
      }
    }
  }
}

// ---------------- fused flash attention per (b,h), 256-query chunks ----------------
__global__ __launch_bounds__(256)
void attn_fused(const _Float16* __restrict__ q16, const _Float16* __restrict__ k16,
                const _Float16* __restrict__ v16, _Float16* __restrict__ ctx16) {
  __shared__ _Float16 kbuf[32][EHEAD];   // 1KB, contiguous
  __shared__ _Float16 vbuf[32][EHEAD];   // 1KB, contiguous
  __shared__ _Float16 stage[8][16][34];  // per-wave P staging (C layout -> A layout)

  const int tid  = threadIdx.x;
  const int wave = tid >> 5, lane = tid & 31, l15 = lane & 15, half = lane >> 4;
  const int bh = blockIdx.x;          // 0..127
  const int b = bh >> 6, h = bh & 63;
  const int qc = blockIdx.y;          // 0..3
  const float scale = 0.03125f;       // 1/sqrt(d_model)

  const _Float16* qh = q16 + (size_t)bh * SEQ * EHEAD;
  const _Float16* kh = k16 + (size_t)bh * SEQ * EHEAD;
  const _Float16* vh = v16 + (size_t)bh * SEQ * EHEAD;

  v16h aq[2];
  v8f  acc[2];
  float mrun[2][8], lrun[2][8];
#pragma unroll
  for (int t = 0; t < 2; ++t) {
    int m0 = qc * 256 + wave * 32 + t * 16;
#pragma unroll
    for (int i = 0; i < 16; ++i) {
      int k = i + 16 * half;
      aq[t][i] = (k < EHEAD)
                     ? (_Float16)((float)qh[(size_t)(m0 + l15) * EHEAD + k] * scale)
                     : (_Float16)0.0f;   // zero-pad K 16..31
    }
#pragma unroll
    for (int v = 0; v < 8; ++v) {
      acc[t][v] = 0.0f; mrun[t][v] = -INFINITY; lrun[t][v] = 0.0f;
    }
  }

  for (int kb = 0; kb < SEQ; kb += 32) {
    // async global->LDS: 32x16 f16 K tile and V tile (1KB each, contiguous)
    if (tid < 64) {
      async_copy16(lds_off(&kbuf[0][0]) + tid * 16,
                   (const char*)(kh + (size_t)kb * EHEAD) + tid * 16);
    } else if (tid < 128) {
      int c = tid - 64;
      async_copy16(lds_off(&vbuf[0][0]) + c * 16,
                   (const char*)(vh + (size_t)kb * EHEAD) + c * 16);
    }
    if (kb + 32 < SEQ && lane == 0) {
      __builtin_prefetch(&kh[(size_t)(kb + 32) * EHEAD], 0, 0);
      __builtin_prefetch(&vh[(size_t)(kb + 32) * EHEAD], 0, 0);
    }
    wait_async0();      // issuing waves drain their ASYNCcnt
    __syncthreads();    // then tiles are visible to every wave

    v16h bk0, bk1, bv;
#pragma unroll
    for (int i = 0; i < 16; ++i) {
      // scores B: B[K=feat][N=key] = k[key][feat]; K>=16 zero-padded
      bk0[i] = half ? (_Float16)0.0f : kbuf[l15][i];
      bk1[i] = half ? (_Float16)0.0f : kbuf[16 + l15][i];
      // ctx B: B[K=key][N=feat] = v[key][feat], full K=32
      bv[i] = vbuf[i + 16 * half][l15];
    }

#pragma unroll
    for (int t = 0; t < 2; ++t) {
      v8f z;
#pragma unroll
      for (int v = 0; v < 8; ++v) z[v] = 0.0f;
      v8f s0 = __builtin_amdgcn_wmma_f32_16x16x32_f16(false, aq[t], false, bk0,
                                                      (short)0, z, false, false);
      v8f s1 = __builtin_amdgcn_wmma_f32_16x16x32_f16(false, aq[t], false, bk1,
                                                      (short)0, z, false, false);
#pragma unroll
      for (int v = 0; v < 8; ++v) {
        float mx   = rowmax16(fmaxf(s0[v], s1[v]));
        float mnew = fmaxf(mrun[t][v], mx);
        float p0 = __expf(s0[v] - mnew);
        float p1 = __expf(s1[v] - mnew);
        float rs = rowsum16(p0 + p1);
        float alpha = __expf(mrun[t][v] - mnew);
        lrun[t][v] = lrun[t][v] * alpha + rs;
        mrun[t][v] = mnew;
        acc[t][v] *= alpha;
        int M = v + 8 * half;
        stage[wave][M][l15]      = (_Float16)p0;
        stage[wave][M][16 + l15] = (_Float16)p1;
      }
      v16h ap;  // P as A fragment (per-wave LDS; DS ops in-order within wave)
#pragma unroll
      for (int i = 0; i < 16; ++i) {
        int k = i + 16 * half;
        ap[i] = stage[wave][l15][k];
      }
      acc[t] = __builtin_amdgcn_wmma_f32_16x16x32_f16(false, ap, false, bv,
                                                      (short)0, acc[t], false, false);
    }
    __syncthreads();
  }

#pragma unroll
  for (int t = 0; t < 2; ++t) {
    int m0 = qc * 256 + wave * 32 + t * 16;
#pragma unroll
    for (int v = 0; v < 8; ++v) {
      int s = m0 + v + 8 * half;
      float o = acc[t][v] / lrun[t][v];
      ctx16[((size_t)(b * SEQ + s)) * D_MODEL + h * EHEAD + l15] = (_Float16)o;
    }
  }
}

extern "C" void kernel_launch(void* const* d_in, const int* in_sizes, int n_in,
                              void* d_out, int out_size, void* d_ws, size_t ws_size,
                              hipStream_t stream) {
  (void)in_sizes; (void)n_in; (void)out_size; (void)ws_size;
  const float* Q  = (const float*)d_in[0];
  const float* K  = (const float*)d_in[1];
  const float* V  = (const float*)d_in[2];
  const float* Wq = (const float*)d_in[3];
  const float* bq = (const float*)d_in[4];
  const float* Wk = (const float*)d_in[5];
  const float* bk = (const float*)d_in[6];
  const float* Wv = (const float*)d_in[7];
  const float* bv = (const float*)d_in[8];
  const float* Wo = (const float*)d_in[9];
  const float* bo = (const float*)d_in[10];
  float* out = (float*)d_out;

  char* ws = (char*)d_ws;
  const size_t WSZ = (size_t)D_MODEL * D_MODEL * sizeof(_Float16);  // 2 MB
  const size_t ACT = (size_t)ROWS * D_MODEL * sizeof(_Float16);     // 4 MB
  _Float16* Wq16 = (_Float16*)(ws + 0 * WSZ);
  _Float16* Wk16 = (_Float16*)(ws + 1 * WSZ);
  _Float16* Wv16 = (_Float16*)(ws + 2 * WSZ);
  _Float16* Wo16 = (_Float16*)(ws + 3 * WSZ);
  char* actbase = ws + 4 * WSZ;                  // 8 MB
  _Float16* X16q = (_Float16*)(actbase + 0 * ACT);
  _Float16* X16k = (_Float16*)(actbase + 1 * ACT);
  _Float16* X16v = (_Float16*)(actbase + 2 * ACT);
  _Float16* q16  = (_Float16*)(actbase + 3 * ACT);
  _Float16* k16  = (_Float16*)(actbase + 4 * ACT);
  _Float16* v16  = (_Float16*)(actbase + 5 * ACT);
  _Float16* c16  = (_Float16*)(actbase + 6 * ACT); // total 36 MB

  const int nW = D_MODEL * D_MODEL;
  const int nX = ROWS * D_MODEL;
  cvt_f32_f16<<<dim3(256), dim3(256), 0, stream>>>(Wq, Wq16, nW);
  cvt_f32_f16<<<dim3(256), dim3(256), 0, stream>>>(Wk, Wk16, nW);
  cvt_f32_f16<<<dim3(256), dim3(256), 0, stream>>>(Wv, Wv16, nW);
  cvt_f32_f16<<<dim3(256), dim3(256), 0, stream>>>(Wo, Wo16, nW);
  cvt_f32_f16<<<dim3(256), dim3(256), 0, stream>>>(Q, X16q, nX);
  cvt_f32_f16<<<dim3(256), dim3(256), 0, stream>>>(K, X16k, nX);
  cvt_f32_f16<<<dim3(256), dim3(256), 0, stream>>>(V, X16v, nX);

  dim3 g(ROWS / 128, D_MODEL / 128), blk(256);
  gemm_wmma_tdm<true><<<g, blk, 0, stream>>>(X16q, Wq16, bq, (void*)q16);
  gemm_wmma_tdm<true><<<g, blk, 0, stream>>>(X16k, Wk16, bk, (void*)k16);
  gemm_wmma_tdm<true><<<g, blk, 0, stream>>>(X16v, Wv16, bv, (void*)v16);

  attn_fused<<<dim3(BATCH * NHEADS, SEQ / 256), blk, 0, stream>>>(q16, k16, v16, c16);

  gemm_wmma_tdm<false><<<g, blk, 0, stream>>>(c16, Wo16, bo, (void*)out);
}
